// PerspectiveNetwork_57672820851425
// MI455X (gfx1250) — matrix operand order repaired and checked by reading, not compiled, not run on hardware
//
#include <hip/hip_runtime.h>

#define B_ROWS 16384
#define F_IN   768
#define H_OUT  1024

typedef __attribute__((ext_vector_type(16))) _Float16 v16h;
typedef __attribute__((ext_vector_type(8)))  _Float16 v8h;
typedef __attribute__((ext_vector_type(8)))  float    v8f;
typedef __attribute__((ext_vector_type(4)))  float    v4f;

// ---------------- prepass: fp32 -> f16 ----------------
__global__ void cvt_f32_to_f16_kernel(const float* __restrict__ in,
                                      _Float16* __restrict__ out, int n) {
  int i = blockIdx.x * blockDim.x + threadIdx.x;
  if (i < n) out[i] = (_Float16)in[i];
}

__global__ void zero_f32_kernel(float* __restrict__ p, int n) {
  int i = blockIdx.x * blockDim.x + threadIdx.x;
  if (i < n) p[i] = 0.0f;
}

// A fragment (16x32 MxK, 16-bit layout) from a pre-converted f16 row:
// two 16-byte (b128) loads of 8 contiguous halves each.
__device__ __forceinline__ v16h a_frag_f16(const _Float16* __restrict__ row,
                                           int kA0, int kA1) {
  v8h lo = *(const v8h*)(row + kA0);
  v8h hi = *(const v8h*)(row + kA1);
  v16h r;
#pragma unroll
  for (int i = 0; i < 8; ++i) { r[i] = lo[i]; r[8 + i] = hi[i]; }
  return r;
}

// A fragment built from fp32 rows with in-register conversion (fallback path
// when the workspace is too small to hold f16 copies of stm/nstm).
__device__ __forceinline__ v16h a_frag_f32(const float* __restrict__ row,
                                           int kA0, int kA1) {
  v4f a = *(const v4f*)(row + kA0);
  v4f b = *(const v4f*)(row + kA0 + 4);
  v4f c = *(const v4f*)(row + kA1);
  v4f d = *(const v4f*)(row + kA1 + 4);
  v16h r;
#pragma unroll
  for (int i = 0; i < 4; ++i) {
    r[i]      = (_Float16)a[i];
    r[4 + i]  = (_Float16)b[i];
    r[8 + i]  = (_Float16)c[i];
    r[12 + i] = (_Float16)d[i];
  }
  return r;
}

// ---------------- main fused kernel ----------------
// grid = (B/64, H/128), block = 256 threads (8 waves of 32).
// Wave (m,n): rows [64*bx + 16m, +16) x cols [128*by + 64n, +64), both
// perspectives. Fused epilogue: bias -> screlu -> dot(out_w) -> per-row
// shuffle reduction -> atomicAdd into acc[b].
template <bool A_IS_F16>
__global__ __launch_bounds__(256, 1)
void ft_gemm_screlu_dot_kernel(const void* __restrict__ stmv,
                               const void* __restrict__ nstmv,
                               const _Float16* __restrict__ ftw,  // [H][F] f16
                               const float* __restrict__ ftb,     // [H]
                               const float* __restrict__ outw,    // [2H]
                               float* __restrict__ acc)           // [B]
{
  const int lane   = threadIdx.x & 31;
  const int wave   = threadIdx.x >> 5;
  const int wave_m = wave & 3;   // 0..3
  const int wave_n = wave >> 2;  // 0..1

  const int bBase = blockIdx.x * 64 + wave_m * 16;
  const int hBase = blockIdx.y * 128 + wave_n * 64;

  const int r  = lane & 15;  // M-row (A) / N-col (B,D) within fragment
  const int hi = lane >> 4;  // half-wave selects K-chunk (A,B) / M+8 (D)

  v8f accS[4] = {};
  v8f accN[4] = {};

  const size_t rowOff = (size_t)(bBase + r) * F_IN;
  const _Float16* aS16 = (const _Float16*)stmv + rowOff;
  const _Float16* aN16 = (const _Float16*)nstmv + rowOff;
  const float*    aS32 = (const float*)stmv + rowOff;
  const float*    aN32 = (const float*)nstmv + rowOff;

  for (int k0 = 0; k0 < F_IN; k0 += 32) {
    // A layout: lanes 0-15 hold K=k0+[0..15] for row M=r;
    // lanes 16-31 hold K=k0+16+[0..15].
    const int kA0 = k0 + hi * 16;
    const int kA1 = k0 + hi * 16 + 8;
    v16h aFs, aFn;
    if constexpr (A_IS_F16) {
      aFs = a_frag_f16(aS16, kA0, kA1);
      aFn = a_frag_f16(aN16, kA0, kA1);
    } else {
      aFs = a_frag_f32(aS32, kA0, kA1);
      aFn = a_frag_f32(aN32, kA0, kA1);
    }

#pragma unroll
    for (int t = 0; t < 4; ++t) {
      // B fragment (32x16 KxN): lane = column N -> h = hBase+16t+r,
      // lanes 0-15 hold K=k0+[0..15], lanes 16-31 hold K=k0+16+[0..15].
      // ft_w is [h][f] row-major f16 -> 16 contiguous halves (2x b128).
      const _Float16* bp =
          ftw + (size_t)(hBase + t * 16 + r) * F_IN + k0 + hi * 16;
      v16h bF = *(const v16h*)bp;

      accS[t] = __builtin_amdgcn_wmma_f32_16x16x32_f16(
          false, aFs, false, bF, (short)0, accS[t], false, false);
      accN[t] = __builtin_amdgcn_wmma_f32_16x16x32_f16(
          false, aFn, false, bF, (short)0, accN[t], false, false);
    }
  }

  // Epilogue: D layout -> VGPR j holds (M = j + 8*hi, N = r).
  float rowPart[8];
#pragma unroll
  for (int j = 0; j < 8; ++j) rowPart[j] = 0.0f;

#pragma unroll
  for (int t = 0; t < 4; ++t) {
    const int h      = hBase + t * 16 + r;
    const float bias = ftb[h];
    const float wS   = outw[h];
    const float wN   = outw[H_OUT + h];
#pragma unroll
    for (int j = 0; j < 8; ++j) {
      float vs = accS[t][j] + bias;
      vs = fminf(fmaxf(vs, 0.0f), 1.0f);
      float vn = accN[t][j] + bias;
      vn = fminf(fmaxf(vn, 0.0f), 1.0f);
      rowPart[j] += vs * vs * wS + vn * vn * wN;
    }
  }

  // Reduce across the 16 N-lanes of each half-wave (xor masks < 16 never
  // cross the lane[4] boundary), then one atomic per row per half-wave.
#pragma unroll
  for (int j = 0; j < 8; ++j) {
    float v = rowPart[j];
    v += __shfl_xor(v, 1);
    v += __shfl_xor(v, 2);
    v += __shfl_xor(v, 4);
    v += __shfl_xor(v, 8);
    if (r == 0) atomicAdd(&acc[bBase + j + hi * 8], v);
  }
}

// ---------------- finalize: + out_b, sigmoid ----------------
__global__ void finalize_kernel(const float* __restrict__ acc,
                                const float* __restrict__ outb,
                                float* __restrict__ out, int n) {
  int i = blockIdx.x * blockDim.x + threadIdx.x;
  if (i < n) {
    float x = acc[i] + outb[0];
    out[i] = 1.0f / (1.0f + __expf(-x));
  }
}

extern "C" void kernel_launch(void* const* d_in, const int* in_sizes, int n_in,
                              void* d_out, int out_size, void* d_ws, size_t ws_size,
                              hipStream_t stream) {
  (void)in_sizes; (void)n_in; (void)out_size;

  const float* stm   = (const float*)d_in[0];  // [B,768]
  const float* nstm  = (const float*)d_in[1];  // [B,768]
  const float* ftw32 = (const float*)d_in[2];  // [1024,768]
  const float* ftb   = (const float*)d_in[3];  // [1024]
  const float* outw  = (const float*)d_in[4];  // [1,2048]
  const float* outb  = (const float*)d_in[5];  // [1]
  float* out = (float*)d_out;                  // [B,1]

  // Workspace layout (256B-aligned chunks):
  //   ftw16 (1.5 MB) | acc (64 KB) | [optional] stm16 (25 MB) | nstm16 (25 MB)
  const size_t ftwBytes = ((size_t)H_OUT * F_IN * sizeof(_Float16) + 255) &
                          ~(size_t)255;
  const size_t accBytes = ((size_t)B_ROWS * sizeof(float) + 255) & ~(size_t)255;
  const size_t aBytes   = ((size_t)B_ROWS * F_IN * sizeof(_Float16) + 255) &
                          ~(size_t)255;

  char* wsp = (char*)d_ws;
  _Float16* ftw16 = (_Float16*)wsp;
  float*    acc   = (float*)(wsp + ftwBytes);
  _Float16* stm16 = (_Float16*)(wsp + ftwBytes + accBytes);
  _Float16* nstm16 = (_Float16*)(wsp + ftwBytes + accBytes + aBytes);

  const bool useF16A = ws_size >= ftwBytes + accBytes + 2 * aBytes;

  const int nW = H_OUT * F_IN;
  cvt_f32_to_f16_kernel<<<(nW + 255) / 256, 256, 0, stream>>>(ftw32, ftw16, nW);
  zero_f32_kernel<<<(B_ROWS + 255) / 256, 256, 0, stream>>>(acc, B_ROWS);

  dim3 grid(B_ROWS / 64, H_OUT / 128);
  if (useF16A) {
    const int nA = B_ROWS * F_IN;
    cvt_f32_to_f16_kernel<<<(nA + 255) / 256, 256, 0, stream>>>(stm, stm16, nA);
    cvt_f32_to_f16_kernel<<<(nA + 255) / 256, 256, 0, stream>>>(nstm, nstm16,
                                                                nA);
    ft_gemm_screlu_dot_kernel<true><<<grid, 256, 0, stream>>>(
        stm16, nstm16, ftw16, ftb, outw, acc);
  } else {
    ft_gemm_screlu_dot_kernel<false><<<grid, 256, 0, stream>>>(
        stm, nstm, ftw16, ftb, outw, acc);
  }

  finalize_kernel<<<(B_ROWS + 255) / 256, 256, 0, stream>>>(acc, outb, out,
                                                            B_ROWS);
}
